// GCN1_84250078479004
// MI455X (gfx1250) — matrix operand order repaired and checked by reading, not compiled, not run on hardware
//
#include <hip/hip_runtime.h>

#define NNODES 100000
#define NEDGES 1600000
#define NFEAT  512
#define NHID   128
#define NCLASS 64

typedef __attribute__((ext_vector_type(16))) __bf16 v16bf;
typedef __attribute__((ext_vector_type(8)))  float  v8f;

union FragAB { uint4 u[2]; v16bf v; };

__device__ __forceinline__ unsigned short f2bf(float f) {
    unsigned int u = __float_as_uint(f);
    u += 0x7FFFu + ((u >> 16) & 1u);        // round-to-nearest-even
    return (unsigned short)(u >> 16);
}

// ---------------------------------------------------------------------------
// Tiled GEMM: out[M x NOUT] = X[M x K] @ W[K x NOUT], bf16 WMMA, f32 accum.
// Block = 256 threads (8 waves). Block tile = 128 rows x NOUT cols.
// Wave w computes rows [w*16, w*16+16) x all NOUT columns.
// ---------------------------------------------------------------------------
template<int K, int NOUT>
__global__ __launch_bounds__(256) void gemm_bf16_wmma(
        const float* __restrict__ X, const float* __restrict__ W,
        float* __restrict__ out, int M)
{
    constexpr int KC = K / 32;       // K chunks
    constexpr int NT = NOUT / 16;    // N tiles per wave

    // padded to 40 shorts (80 B): conflict-free strides, 16B-aligned rows
    __shared__ __align__(16) unsigned short sX[128][40];
    __shared__ __align__(16) unsigned short sWT[NOUT][40];   // W transposed [n][k]

    const int t    = threadIdx.x;
    const int lane = t & 31;
    const int wid  = t >> 5;
    const int m0   = blockIdx.x * 128;

    v8f acc[NT] = {};

    // ISA 7.12.2 fragment coordinates (wave32)
    const int rowA  = (wid << 4) + (lane & 15);   // A row in block tile
    const int kbase = (lane >> 4) << 3;           // A: K 0..7/16..23 vs 8..15/24..31
    const int colB  = lane & 15;                  // B column
    const int ksB   = (lane >> 4) << 3;           // B: K 0..7/16..23 vs 8..15/24..31

    for (int kc = 0; kc < KC; ++kc) {
        __syncthreads();
        // stage X tile (128 rows x 32 k) fp32 -> bf16; coalesced float4 reads
        for (int i = 0; i < 4; ++i) {
            int idx = t + i * 256;                 // 1024 float4 slots
            int r   = idx >> 3;
            int c4  = (idx & 7) << 2;
            float4 v = make_float4(0.f, 0.f, 0.f, 0.f);
            int gr = m0 + r;
            if (gr < M)
                v = *(const float4*)&X[(size_t)gr * K + kc * 32 + c4];
            sX[r][c4 + 0] = f2bf(v.x);
            sX[r][c4 + 1] = f2bf(v.y);
            sX[r][c4 + 2] = f2bf(v.z);
            sX[r][c4 + 3] = f2bf(v.w);
        }
        // stage W tile transposed (NOUT x 32); coalesced float4 reads of W rows
        constexpr int WELEM4 = 32 * NOUT / 4;
        for (int i = 0; i < WELEM4 / 256; ++i) {
            int idx = t + i * 256;
            int k   = idx / (NOUT / 4);
            int n4  = (idx % (NOUT / 4)) << 2;
            float4 v = *(const float4*)&W[(size_t)(kc * 32 + k) * NOUT + n4];
            sWT[n4 + 0][k] = f2bf(v.x);
            sWT[n4 + 1][k] = f2bf(v.y);
            sWT[n4 + 2][k] = f2bf(v.z);
            sWT[n4 + 3][k] = f2bf(v.w);
        }
        __syncthreads();

        FragAB a;
        a.u[0] = *(const uint4*)&sX[rowA][kbase];        // K kbase..kbase+7
        a.u[1] = *(const uint4*)&sX[rowA][kbase + 16];   // K kbase+16..kbase+23
        for (int nt = 0; nt < NT; ++nt) {
            FragAB b;
            b.u[0] = *(const uint4*)&sWT[nt * 16 + colB][ksB];
            b.u[1] = *(const uint4*)&sWT[nt * 16 + colB][ksB + 16];
            acc[nt] = __builtin_amdgcn_wmma_f32_16x16x32_bf16(
                false, a.v, false, b.v, (short)0, acc[nt], false, false);
        }
    }

    // C/D layout: element r -> M = (lane>>4)*8 + r, N = lane&15
    const int mW = m0 + (wid << 4) + ((lane >> 4) << 3);
    const int nL = lane & 15;
    for (int nt = 0; nt < NT; ++nt)
        for (int r = 0; r < 8; ++r) {
            int row = mW + r;
            if (row < M) out[(size_t)row * NOUT + nt * 16 + nL] = acc[nt][r];
        }
}

// ---------------------------------------------------------------------------
// Edge aggregation: agg[dst] += support[src] * w  (COO scatter-add).
// F/4 consecutive threads cover one edge's feature row -> coalesced bursts.
// Each thread slot processes 4 consecutive edges: all four random-row gathers
// are issued before any atomic, so 4 float4 loads are in flight per thread.
// NEDGES is a multiple of 4, so slot count is exact (no tail guard needed).
// ---------------------------------------------------------------------------
template<int F>
__global__ void edge_scatter(const float* __restrict__ sup,
                             const int* __restrict__ src,
                             const int* __restrict__ dst,
                             const float* __restrict__ ew,
                             float* __restrict__ agg)
{
    constexpr int PER = F / 4;                       // threads per edge
    long long tid = (long long)blockIdx.x * blockDim.x + threadIdx.x;
    long long q = tid / PER;                         // edge-group slot
    long long e0 = q * 4;
    if (e0 >= NEDGES) return;
    int f = (int)(tid % PER) << 2;

    // prefetch upcoming edge-stream cachelines (sequential streams)
    __builtin_prefetch(&src[e0 + 64], 0, 0);
    __builtin_prefetch(&dst[e0 + 64], 0, 0);
    __builtin_prefetch(&ew[e0 + 64], 0, 0);

    // edge metadata: 16B coalesced vector loads
    int4   s4 = *(const int4*)&src[e0];
    int4   d4 = *(const int4*)&dst[e0];
    float4 w4 = *(const float4*)&ew[e0];

    // issue all four gathers up front (independent -> 4 loads in flight)
    float4 v0 = *(const float4*)&sup[(size_t)s4.x * F + f];
    float4 v1 = *(const float4*)&sup[(size_t)s4.y * F + f];
    float4 v2 = *(const float4*)&sup[(size_t)s4.z * F + f];
    float4 v3 = *(const float4*)&sup[(size_t)s4.w * F + f];

    float* p0 = &agg[(size_t)d4.x * F + f];
    unsafeAtomicAdd(p0 + 0, v0.x * w4.x);
    unsafeAtomicAdd(p0 + 1, v0.y * w4.x);
    unsafeAtomicAdd(p0 + 2, v0.z * w4.x);
    unsafeAtomicAdd(p0 + 3, v0.w * w4.x);

    float* p1 = &agg[(size_t)d4.y * F + f];
    unsafeAtomicAdd(p1 + 0, v1.x * w4.y);
    unsafeAtomicAdd(p1 + 1, v1.y * w4.y);
    unsafeAtomicAdd(p1 + 2, v1.z * w4.y);
    unsafeAtomicAdd(p1 + 3, v1.w * w4.y);

    float* p2 = &agg[(size_t)d4.z * F + f];
    unsafeAtomicAdd(p2 + 0, v2.x * w4.z);
    unsafeAtomicAdd(p2 + 1, v2.y * w4.z);
    unsafeAtomicAdd(p2 + 2, v2.z * w4.z);
    unsafeAtomicAdd(p2 + 3, v2.w * w4.z);

    float* p3 = &agg[(size_t)d4.w * F + f];
    unsafeAtomicAdd(p3 + 0, v3.x * w4.w);
    unsafeAtomicAdd(p3 + 1, v3.y * w4.w);
    unsafeAtomicAdd(p3 + 2, v3.z * w4.w);
    unsafeAtomicAdd(p3 + 3, v3.w * w4.w);
}

__global__ void zero_f32(float* __restrict__ p, long long n)
{
    long long i = (long long)blockIdx.x * blockDim.x + threadIdx.x;
    if (i < n) p[i] = 0.0f;
}

__global__ void bias_lrelu(const float* __restrict__ agg,
                           const float* __restrict__ b,
                           float* __restrict__ h)
{
    long long i = (long long)blockIdx.x * blockDim.x + threadIdx.x;
    if (i >= (long long)NNODES * NHID) return;
    float x = agg[i] + b[i & (NHID - 1)];
    h[i] = x > 0.0f ? x : 0.01f * x;
}

// one wave per row of 64 logits; 2 elements per lane; shfl reductions (wave32)
__global__ __launch_bounds__(256) void softmax64(
        const float* __restrict__ logits, const float* __restrict__ b,
        float* __restrict__ out)
{
    int wid = threadIdx.x >> 5, lane = threadIdx.x & 31;
    int row = blockIdx.x * 8 + wid;
    if (row >= NNODES) return;
    size_t base = (size_t)row * NCLASS;
    float x0 = logits[base + lane]      + b[lane];
    float x1 = logits[base + lane + 32] + b[lane + 32];
    float m = fmaxf(x0, x1);
    for (int i = 16; i > 0; i >>= 1) m = fmaxf(m, __shfl_xor(m, i, 32));
    float e0 = __expf(x0 - m), e1 = __expf(x1 - m);
    float s = e0 + e1;
    for (int i = 16; i > 0; i >>= 1) s += __shfl_xor(s, i, 32);
    float inv = 1.0f / s;
    out[base + lane]      = e0 * inv;
    out[base + lane + 32] = e1 * inv;
}

extern "C" void kernel_launch(void* const* d_in, const int* in_sizes, int n_in,
                              void* d_out, int out_size, void* d_ws, size_t ws_size,
                              hipStream_t stream) {
    const float* y  = (const float*)d_in[0];
    const int*   src = (const int*)d_in[1];
    const int*   dst = (const int*)d_in[2];
    const float* ew = (const float*)d_in[3];
    const float* W1 = (const float*)d_in[4];
    const float* b1 = (const float*)d_in[5];
    const float* W2 = (const float*)d_in[6];
    const float* b2 = (const float*)d_in[7];

    float* softout = (float*)d_out;                       // [N, 64]
    float* h       = softout + (size_t)NNODES * NCLASS;   // [N, 128] embedding
    float* sup     = (float*)d_ws;                        // [N, 128] (reused)
    float* agg     = sup + (size_t)NNODES * NHID;         // [N, 128] (reused)

    const int gblk = (NNODES + 127) / 128;
    const long long n1 = (long long)NNODES * NHID;
    const long long n2 = (long long)NNODES * NCLASS;

    // Layer 1
    gemm_bf16_wmma<NFEAT, NHID><<<gblk, 256, 0, stream>>>(y, W1, sup, NNODES);
    zero_f32<<<(int)((n1 + 255) / 256), 256, 0, stream>>>(agg, n1);
    long long t1 = (long long)(NEDGES / 4) * (NHID / 4);  // slots * threads-per-edge
    edge_scatter<NHID><<<(int)((t1 + 255) / 256), 256, 0, stream>>>(sup, src, dst, ew, agg);
    bias_lrelu<<<(int)((n1 + 255) / 256), 256, 0, stream>>>(agg, b1, h);

    // Layer 2
    gemm_bf16_wmma<NHID, NCLASS><<<gblk, 256, 0, stream>>>(h, W2, sup, NNODES);
    zero_f32<<<(int)((n2 + 255) / 256), 256, 0, stream>>>(agg, n2);
    long long t2 = (long long)(NEDGES / 4) * (NCLASS / 4);
    edge_scatter<NCLASS><<<(int)((t2 + 255) / 256), 256, 0, stream>>>(sup, src, dst, ew, agg);
    softmax64<<<(NNODES + 7) / 8, 256, 0, stream>>>(agg, b2, softout);
}
